// SynthesisNetwork_75256416961042
// MI455X (gfx1250) — matrix operand order repaired and checked by reading, not compiled, not run on hardware
//
#include <hip/hip_runtime.h>
#include <hip/hip_bf16.h>
#include <math.h>

// Problem dims
#define B_    64
#define T_    1200
#define U_    80
#define V_    60
#define H_    400
#define KMIX  20
#define BT    (B_ * T_)     // 76800 rows
#define G4H   1600          // 4*H
#define KP_H  416           // H padded to mult-of-32 (K dim for h1 / W_att / W_hh2)
#define KP_R2 480           // 463 (3+V+H) padded
#define KP_LIN 800          // 2H (already mult of 32)
#define N_ATT 64            // 3K=60 padded
#define N_OUT 128           // 6K+1=121 padded
#define NB_SCAN 25          // persistent scan blocks (25*4 waves = 100 wave-jobs)

typedef __bf16 bf16_t;
typedef __bf16 v16bf __attribute__((ext_vector_type(16)));
typedef float  v8f   __attribute__((ext_vector_type(8)));

__device__ __forceinline__ bf16_t f2bf(float f) {
  unsigned u = __float_as_uint(f);
  u += 0x7FFFu + ((u >> 16) & 1u);              // round-to-nearest-even
  unsigned short h = (unsigned short)(u >> 16);
  return __builtin_bit_cast(bf16_t, h);
}
__device__ __forceinline__ float sigf(float x) { return 1.0f / (1.0f + expf(-x)); }

// ---- WMMA tile loaders (v_wmma_f32_16x16x32_bf16 operand layouts) ----
// A operand, 16x32 bf16: lane -> M = lane&15; half = lane>>4 selects contiguous
// 16-K block; VGPRs step K by 2 (same K order as B).
__device__ __forceinline__ v16bf load_tile_a(const bf16_t* tile, int ld, int lane) {
  const bf16_t* row = tile + (size_t)(lane & 15) * ld + ((lane >> 4) << 4);
  union { v16bf v; unsigned u[8]; } r;
#pragma unroll
  for (int i = 0; i < 8; ++i) {
    int k = 2 * i; // 0,2,...,14
    r.u[i] = *reinterpret_cast<const unsigned*>(row + k);
  }
  return r.v;
}
// B operand, 32x16 bf16 (we feed W as N x K row-major, i.e. B = W^T):
// lane -> N = lane&15; half selects contiguous 16-K block; VGPRs step K by 2.
__device__ __forceinline__ v16bf load_tile_b(const bf16_t* tile, int ld, int lane) {
  const bf16_t* row = tile + (size_t)(lane & 15) * ld + ((lane >> 4) << 4);
  union { v16bf v; unsigned u[8]; } r;
#pragma unroll
  for (int i = 0; i < 8; ++i)
    r.u[i] = *reinterpret_cast<const unsigned*>(row + 2 * i);
  return r.v;
}

// 16x64 output block, fully unrolled K (compile-time): one A fragment per
// K-step feeds 4 in-place WMMAs. Named accumulators + full unroll avoid the
// pipeliner's accumulator-rotation copies (and their WMMA->VALU hazard NOPs).
template <int KT>
__device__ __forceinline__ void wmma_gemm_x4(const bf16_t* __restrict__ A, int lda,
                                             const bf16_t* __restrict__ W, int ldw,
                                             int lane, v8f& acc0, v8f& acc1,
                                             v8f& acc2, v8f& acc3) {
  const bf16_t* W0 = W;
  const bf16_t* W1 = W + (size_t)16 * ldw;
  const bf16_t* W2 = W + (size_t)32 * ldw;
  const bf16_t* W3 = W + (size_t)48 * ldw;
#pragma unroll
  for (int k = 0; k < KT; k += 32) {
    v16bf a  = load_tile_a(A + k, lda, lane);
    v16bf b0 = load_tile_b(W0 + k, ldw, lane);
    v16bf b1 = load_tile_b(W1 + k, ldw, lane);
    v16bf b2 = load_tile_b(W2 + k, ldw, lane);
    v16bf b3 = load_tile_b(W3 + k, ldw, lane);
    acc0 = __builtin_amdgcn_wmma_f32_16x16x32_bf16(false, a, false, b0, (short)0, acc0, false, false);
    acc1 = __builtin_amdgcn_wmma_f32_16x16x32_bf16(false, a, false, b1, (short)0, acc1, false, false);
    acc2 = __builtin_amdgcn_wmma_f32_16x16x32_bf16(false, a, false, b2, (short)0, acc2, false, false);
    acc3 = __builtin_amdgcn_wmma_f32_16x16x32_bf16(false, a, false, b3, (short)0, acc3, false, false);
  }
}

// ---- weight conversion / padding (f32 -> bf16, zero pad) ----
__global__ void k_prep(const float* __restrict__ Watt, const float* __restrict__ Wih2,
                       const float* __restrict__ Whh2, const float* __restrict__ Wout,
                       bf16_t* __restrict__ Wattb, bf16_t* __restrict__ Wih2b,
                       bf16_t* __restrict__ Whh2b, bf16_t* __restrict__ Woutb) {
  int gid = blockIdx.x * blockDim.x + threadIdx.x;
  const int R0 = N_ATT * KP_H;         // 26624
  const int R1 = R0 + G4H * KP_R2;     // +768000
  const int R2 = R1 + G4H * KP_H;      // +665600
  const int R3 = R2 + N_OUT * KP_LIN;  // +102400 = 1562624
  if (gid < R0) {
    int n = gid / KP_H, k = gid % KP_H;
    float v = (n < 60 && k < H_) ? Watt[n * H_ + k] : 0.f;
    Wattb[gid] = f2bf(v);
  } else if (gid < R1) {
    int i = gid - R0, n = i / KP_R2, k = i % KP_R2;
    float v = (k < 463) ? Wih2[n * 463 + k] : 0.f;
    Wih2b[i] = f2bf(v);
  } else if (gid < R2) {
    int i = gid - R1, n = i / KP_H, k = i % KP_H;
    float v = (k < H_) ? Whh2[n * H_ + k] : 0.f;
    Whh2b[i] = f2bf(v);
  } else if (gid < R3) {
    int i = gid - R2, n = i / KP_LIN, k = i % KP_LIN;
    float v = (n < 121) ? Wout[n * KP_LIN + k] : 0.f;
    Woutb[i] = f2bf(v);
  }
}

// ---- "LSTM1": gates1 from 3 stroke features only (V cols are zeros; no recurrence) ----
__global__ void k_front(const float* __restrict__ strokes, const float* __restrict__ W_ih1,
                        const float* __restrict__ b_ih1, const float* __restrict__ b_hh1,
                        bf16_t* __restrict__ h1b, bf16_t* __restrict__ rnn2_in,
                        bf16_t* __restrict__ lin_in) {
  const int r = blockIdx.x;
  const float s0 = strokes[r * 3 + 0], s1 = strokes[r * 3 + 1], s2 = strokes[r * 3 + 2];
  bf16_t* h1row = h1b + (size_t)r * KP_H;
  bf16_t* r2row = rnn2_in + (size_t)r * KP_R2;
  bf16_t* lirow = lin_in + (size_t)r * KP_LIN;
  for (int h = threadIdx.x; h < H_; h += blockDim.x) {
    const int ji = h, jg = 2 * H_ + h, jo = 3 * H_ + h;
    float iv = s0 * W_ih1[ji * 63 + 0] + s1 * W_ih1[ji * 63 + 1] + s2 * W_ih1[ji * 63 + 2] + b_ih1[ji] + b_hh1[ji];
    float gv = s0 * W_ih1[jg * 63 + 0] + s1 * W_ih1[jg * 63 + 1] + s2 * W_ih1[jg * 63 + 2] + b_ih1[jg] + b_hh1[jg];
    float ov = s0 * W_ih1[jo * 63 + 0] + s1 * W_ih1[jo * 63 + 1] + s2 * W_ih1[jo * 63 + 2] + b_ih1[jo] + b_hh1[jo];
    float c1 = sigf(iv) * tanhf(gv);
    float h1 = sigf(ov) * tanhf(c1);
    h1row[h] = f2bf(h1);
    r2row[63 + h] = f2bf(c1);   // rnn2_in = [strokes | attention | c1 | pad]
    lirow[h] = f2bf(c1);        // lin_in  = [c1 | rnn2_out]
  }
  if (threadIdx.x < 16) h1row[H_ + threadIdx.x] = f2bf(0.f);       // pad 400..415
  if (threadIdx.x < 17) r2row[463 + threadIdx.x] = f2bf(0.f);      // pad 463..479
  if (threadIdx.x < 3)  r2row[threadIdx.x] = f2bf(strokes[r * 3 + threadIdx.x]);
}

// ---- p = h1 @ W_att^T + b_att : 76800 x 60, K=416; one wave covers N=64 ----
__global__ void __launch_bounds__(256) k_gemm_p(const bf16_t* __restrict__ h1b,
                                                const bf16_t* __restrict__ Wattb,
                                                const float* __restrict__ b_att,
                                                float* __restrict__ p) {
  const int lane = threadIdx.x & 31;
  const int tm = (blockIdx.x * blockDim.x + threadIdx.x) >> 5;   // 4800 waves
  v8f a0 = {}, a1 = {}, a2 = {}, a3 = {};
  wmma_gemm_x4<KP_H>(h1b + (size_t)tm * 16 * KP_H, KP_H, Wattb, KP_H, lane, a0, a1, a2, a3);
  v8f acc[4] = {a0, a1, a2, a3};
  const int mb = (lane >> 4) << 3;
#pragma unroll
  for (int j = 0; j < 4; ++j) {
    const int col = j * 16 + (lane & 15);
    if (col < 60) {
      const float bv = b_att[col];
#pragma unroll
      for (int rI = 0; rI < 8; ++rI)
        p[(size_t)(tm * 16 + mb + rI) * 60 + col] = acc[j][rI] + bv;
    }
  }
}

// ---- attention window + context; writes attention_out (output 0) and rnn2_in cols 3..62 ----
__global__ void k_att(const float* __restrict__ p, const float* __restrict__ sent_oh,
                      const float* __restrict__ sent_mask, float* __restrict__ att_out,
                      bf16_t* __restrict__ rnn2_in) {
  __shared__ float al[KMIX], be[KMIX], ka[KMIX], phi[U_];
  const int r = blockIdx.x;
  const int b = r / T_;
  const int tid = threadIdx.x;
  if (tid < 60) {
    float e = expf(p[(size_t)r * 60 + tid]);
    if (tid < 20) al[tid] = e;
    else if (tid < 40) be[tid - 20] = e;
    else ka[tid - 40] = e;
  }
  __syncthreads();
  if (tid < U_) {
    float s = 0.f;
#pragma unroll
    for (int k = 0; k < KMIX; ++k) {
      float d = ka[k] - (float)(tid + 1);
      s += al[k] * expf(-be[k] * d * d);
    }
    phi[tid] = s * sent_mask[b * U_ + tid];
  }
  __syncthreads();
  if (tid < V_) {
    const float* sb = sent_oh + (size_t)b * U_ * V_ + tid;
    float s = 0.f;
    for (int u = 0; u < U_; ++u) s += phi[u] * sb[u * V_];
    att_out[(size_t)r * V_ + tid] = s;
    rnn2_in[(size_t)r * KP_R2 + 3 + tid] = f2bf(s);
  }
}

// ---- xg = rnn2_in @ W_ih2^T + (b_ih2+b_hh2) : 76800 x 1600, K=480 ----
// 4-way N-blocked: 4800 M-tiles x 25 N-groups = 120000 waves.
__global__ void __launch_bounds__(256) k_gemm_xg(const bf16_t* __restrict__ X,
                                                 const bf16_t* __restrict__ W,
                                                 const float* __restrict__ b_ih2,
                                                 const float* __restrict__ b_hh2,
                                                 float* __restrict__ xg) {
  const int lane = threadIdx.x & 31;
  const int wid = (blockIdx.x * blockDim.x + threadIdx.x) >> 5;
  const int tm = wid / 25, tng = wid % 25;        // N-group of 4 tiles (64 cols)
  v8f a0 = {}, a1 = {}, a2 = {}, a3 = {};
  wmma_gemm_x4<KP_R2>(X + (size_t)tm * 16 * KP_R2, KP_R2,
                      W + (size_t)tng * 64 * KP_R2, KP_R2, lane, a0, a1, a2, a3);
  v8f acc[4] = {a0, a1, a2, a3};
  const int mb = (lane >> 4) << 3;
#pragma unroll
  for (int j = 0; j < 4; ++j) {
    const int col = tng * 64 + j * 16 + (lane & 15);
    const float bv = b_ih2[col] + b_hh2[col];
#pragma unroll
    for (int rI = 0; rI < 8; ++rI)
      xg[(size_t)(tm * 16 + mb + rI) * G4H + col] = acc[j][rI] + bv;
  }
}

// ---- grid barrier: monotonic counter (reset by memset each call) ----
__device__ __forceinline__ void grid_bar(unsigned* cnt, unsigned nb) {
  __syncthreads();
  if (threadIdx.x == 0) {
    __threadfence();
    unsigned my = atomicAdd(cnt, 1u);
    unsigned target = (my / nb + 1u) * nb;
    while (atomicAdd(cnt, 0u) < target) __builtin_amdgcn_s_sleep(1);
    __threadfence();
  }
  __syncthreads();
}

// ---- persistent recurrent LSTM2 scan: per step gates = xg_t + h @ W_hh2^T ----
// 25 blocks x 4 waves = 100 wave-jobs (tm in 0..3, tng in 0..24).
__global__ void __launch_bounds__(128) k_scan(const float* __restrict__ xg,
                                              const bf16_t* __restrict__ Whh2b,
                                              float* __restrict__ gates,   // 64 x 1600
                                              float* __restrict__ cst,     // 64 x 400
                                              bf16_t* __restrict__ hst,    // 64 x 416 (pad zeroed)
                                              bf16_t* __restrict__ lin_in,
                                              unsigned* __restrict__ cnt) {
  const int lane = threadIdx.x & 31;
  const int w = blockIdx.x * 4 + (threadIdx.x >> 5);  // 100 wave-jobs = 4 x 25
  const int tm = w / 25, tng = w % 25;
  const int mb = (lane >> 4) << 3;
  const int gtid = blockIdx.x * 128 + threadIdx.x;    // 3200 threads for update
  const bf16_t* Ahst = hst + (size_t)tm * 16 * KP_H;
  const bf16_t* Wn = Whh2b + (size_t)tng * 64 * KP_H;
  for (int t = 0; t < T_; ++t) {
    v8f a0 = {}, a1 = {}, a2 = {}, a3 = {};
    wmma_gemm_x4<KP_H>(Ahst, KP_H, Wn, KP_H, lane, a0, a1, a2, a3);
    v8f acc[4] = {a0, a1, a2, a3};
#pragma unroll
    for (int j = 0; j < 4; ++j) {
      const int col = tng * 64 + j * 16 + (lane & 15);
#pragma unroll
      for (int rI = 0; rI < 8; ++rI) {
        int bb = tm * 16 + mb + rI;
        gates[bb * G4H + col] = acc[j][rI] + xg[((size_t)bb * T_ + t) * G4H + col];
      }
    }
    grid_bar(cnt, NB_SCAN);
    for (int idx = gtid; idx < B_ * H_; idx += NB_SCAN * 128) {
      int bb = idx / H_, j = idx - bb * H_;
      const float* gr = gates + bb * G4H;
      float iv = gr[j], fv = gr[H_ + j], gv = gr[2 * H_ + j], ov = gr[3 * H_ + j];
      float c = sigf(fv) * cst[idx] + sigf(iv) * tanhf(gv);
      float h = sigf(ov) * tanhf(c);
      cst[idx] = c;
      hst[bb * KP_H + j] = f2bf(h);
      lin_in[((size_t)bb * T_ + t) * KP_LIN + H_ + j] = f2bf(h);
    }
    grid_bar(cnt, NB_SCAN);
  }
}

// ---- lin_out = lin_in @ W_out^T + b_out, scattered into d_out heads ----
// N=128 -> 2 N-groups of 4 tiles; waves = 4800 * 2.
__global__ void __launch_bounds__(256) k_gemm_out(const bf16_t* __restrict__ lin_in,
                                                  const bf16_t* __restrict__ Woutb,
                                                  const float* __restrict__ b_out,
                                                  float* __restrict__ dout) {
  const int lane = threadIdx.x & 31;
  const int wid = (blockIdx.x * blockDim.x + threadIdx.x) >> 5;
  const int tm = wid >> 1, tng = wid & 1;
  v8f a0 = {}, a1 = {}, a2 = {}, a3 = {};
  wmma_gemm_x4<KP_LIN>(lin_in + (size_t)tm * 16 * KP_LIN, KP_LIN,
                       Woutb + (size_t)tng * 64 * KP_LIN, KP_LIN, lane, a0, a1, a2, a3);
  v8f acc[4] = {a0, a1, a2, a3};
  const int mb = (lane >> 4) << 3;
  const size_t OFF_E = (size_t)BT * V_;         // after attention_out
  const size_t OFF_PI = OFF_E + BT;             // after e_hat
  const size_t STR = (size_t)BT * KMIX;         // per-head stride
#pragma unroll
  for (int j = 0; j < 4; ++j) {
    const int col = tng * 64 + j * 16 + (lane & 15);
    if (col < 121) {
      const float bv = b_out[col];
#pragma unroll
      for (int rI = 0; rI < 8; ++rI) {
        int m = tm * 16 + mb + rI;
        float v = acc[j][rI] + bv;
        if (col < 120) {
          int g = col / KMIX, k = col - g * KMIX;   // pi,mu1,mu2,s1,s2,rho
          dout[OFF_PI + (size_t)g * STR + (size_t)m * KMIX + k] = v;
        } else {
          dout[OFF_E + m] = v;                      // e_hat
        }
      }
    }
  }
}

extern "C" void kernel_launch(void* const* d_in, const int* in_sizes, int n_in,
                              void* d_out, int out_size, void* d_ws, size_t ws_size,
                              hipStream_t stream) {
  const float* strokes   = (const float*)d_in[0];
  const float* sent_oh   = (const float*)d_in[1];
  const float* sent_mask = (const float*)d_in[2];
  const float* W_ih1     = (const float*)d_in[3];
  const float* b_ih1     = (const float*)d_in[4];
  const float* b_hh1     = (const float*)d_in[5];
  const float* W_att     = (const float*)d_in[6];
  const float* b_att     = (const float*)d_in[7];
  const float* W_ih2     = (const float*)d_in[8];
  const float* W_hh2     = (const float*)d_in[9];
  const float* b_ih2     = (const float*)d_in[10];
  const float* b_hh2     = (const float*)d_in[11];
  const float* W_out     = (const float*)d_in[12];
  const float* b_out     = (const float*)d_in[13];
  float* dout = (float*)d_out;
  (void)in_sizes; (void)n_in; (void)out_size; (void)ws_size;

  // Workspace carve-up (~770 MB total, dominated by xg f32 491 MB)
  char* ws = (char*)d_ws;
  size_t off = 0;
  auto take = [&](size_t bytes) -> char* {
    char* pp = ws + off;
    off = (off + bytes + 255) & ~(size_t)255;
    return pp;
  };
  unsigned* cnt   = (unsigned*)take(256);
  bf16_t* Wattb   = (bf16_t*)take((size_t)N_ATT * KP_H * 2);
  bf16_t* Wih2b   = (bf16_t*)take((size_t)G4H * KP_R2 * 2);
  bf16_t* Whh2b   = (bf16_t*)take((size_t)G4H * KP_H * 2);
  bf16_t* Woutb   = (bf16_t*)take((size_t)N_OUT * KP_LIN * 2);
  bf16_t* h1b     = (bf16_t*)take((size_t)BT * KP_H * 2);
  bf16_t* rnn2_in = (bf16_t*)take((size_t)BT * KP_R2 * 2);
  bf16_t* lin_in  = (bf16_t*)take((size_t)BT * KP_LIN * 2);
  float*  pbuf    = (float*)take((size_t)BT * 60 * 4);
  float*  xg      = (float*)take((size_t)BT * G4H * 4);
  float*  gates   = (float*)take((size_t)B_ * G4H * 4);
  float*  cst     = (float*)take((size_t)B_ * H_ * 4);
  bf16_t* hst     = (bf16_t*)take((size_t)B_ * KP_H * 2);

  k_prep<<<(1562624 + 255) / 256, 256, 0, stream>>>(W_att, W_ih2, W_hh2, W_out,
                                                    Wattb, Wih2b, Whh2b, Woutb);
  k_front<<<BT, 128, 0, stream>>>(strokes, W_ih1, b_ih1, b_hh1, h1b, rnn2_in, lin_in);
  k_gemm_p<<<(BT / 16) / 8, 256, 0, stream>>>(h1b, Wattb, b_att, pbuf);            // 600 blocks
  k_att<<<BT, 128, 0, stream>>>(pbuf, sent_oh, sent_mask, dout, rnn2_in);
  k_gemm_xg<<<(BT / 16) * 25 / 8, 256, 0, stream>>>(rnn2_in, Wih2b, b_ih2, b_hh2, xg); // 15000 blocks
  hipMemsetAsync(cnt, 0, 256, stream);                         // barrier counter
  hipMemsetAsync(cst, 0, (size_t)B_ * H_ * 4, stream);         // c0 = 0
  hipMemsetAsync(hst, 0, (size_t)B_ * KP_H * 2, stream);       // h0 = 0 (+pad)
  k_scan<<<NB_SCAN, 128, 0, stream>>>(xg, Whh2b, gates, cst, hst, lin_in, cnt);
  k_gemm_out<<<(BT / 16) * 2 / 8, 256, 0, stream>>>(lin_in, Woutb, b_out, dout);   // 1200 blocks
}
